// GraphAttention_73933567033967
// MI455X (gfx1250) — matrix operand order repaired and checked by reading, not compiled, not run on hardware
//
#include <hip/hip_runtime.h>
#include <hip/hip_bf16.h>
#include <math.h>

#define N_NODES   50000
#define N_EDGES   1600000
#define E_TOT     (N_EDGES + N_NODES)   // self-loops appended
#define IN_F      2000
#define OUTF      64
#define NEG_SLOPE 0.2f
#define BN_EPS    1e-5f

typedef __attribute__((ext_vector_type(16))) __bf16 v16bf;
typedef __attribute__((ext_vector_type(8)))  float  v8f;

// ---------- helpers ----------
__device__ __forceinline__ __bf16 f2bf(float f) {
  unsigned u = __builtin_bit_cast(unsigned, f);
  unsigned r = (u + 0x7FFFu + ((u >> 16) & 1u)) >> 16;
  unsigned short s = (unsigned short)r;
  return __builtin_bit_cast(__bf16, s);
}
// monotone key so unsigned atomicMax == float max
__device__ __forceinline__ unsigned fkey(float f) {
  unsigned b = __builtin_bit_cast(unsigned, f);
  return (b & 0x80000000u) ? ~b : (b | 0x80000000u);
}
__device__ __forceinline__ float funkey(unsigned k) {
  unsigned b = (k & 0x80000000u) ? (k & 0x7FFFFFFFu) : ~k;
  return __builtin_bit_cast(float, b);
}
__device__ __forceinline__ void edge_pair(const int* __restrict__ ei, int e, int& s, int& d) {
  if (e < N_EDGES) { s = ei[e]; d = ei[N_EDGES + e]; }
  else             { s = e - N_EDGES; d = s; }           // self loop
}

// ---------- zero fill ----------
__global__ void zero_kernel(float* __restrict__ p, long long n) {
  long long gid = (long long)blockIdx.x * blockDim.x + threadIdx.x;
  if (gid < n) p[gid] = 0.0f;
}

// ---------- bf16 WMMA GEMM: out[N,64] = A[N,K] @ W[K,64] ----------
// BM=64 rows per block, 128 threads (4 waves; wave w owns col tile 16w..16w+15,
// and iterates 4 row sub-tiles sharing one B fragment per WMMA K-step).
#define BM     64
#define KCHUNK 128
#define LDA    (KCHUNK + 8)   // bf16 elems; 272B row stride (bank-conflict pad)
__global__ __launch_bounds__(128)
void gemm_bf16_wmma(const float* __restrict__ A, const float* __restrict__ W,
                    float* __restrict__ out, int K) {
  __shared__ __bf16 As[BM * LDA];   // A tile  64 x KCHUNK
  __shared__ __bf16 Bs[64 * LDA];   // W tile transposed: Bs[n][kk] = W[k0+kk][n]
  const int t    = threadIdx.x;
  const int lane = t & 31;
  const int wave = t >> 5;
  const int row0 = blockIdx.x * BM;

  v8f acc0 = {}, acc1 = {}, acc2 = {}, acc3 = {};

  const int m    = lane & 15;
  const int hi   = lane >> 4;             // half-wave select
  const int ncol = (wave << 4) | (lane & 15);

  for (int k0 = 0; k0 < K; k0 += KCHUNK) {
    if (k0 + KCHUNK <= K) {
      // ---- fast staging (no per-element guards) ----
      // A: 64 rows x 128 k; float4 loads -> 4x bf16 (b64 LDS store)
      #pragma unroll
      for (int it = 0; it < 16; ++it) {
        int r = it * 4 + (t >> 5);
        int c = (t & 31) * 4;
        int rg = row0 + r; if (rg > N_NODES - 1) rg = N_NODES - 1;  // clamp: unused rows
        const float4 v = *(const float4*)&A[(size_t)rg * K + k0 + c];
        __bf16* dp = &As[r * LDA + c];
        dp[0] = f2bf(v.x); dp[1] = f2bf(v.y); dp[2] = f2bf(v.z); dp[3] = f2bf(v.w);
      }
      // B: 128 k x 64 n, transposed; 8 strided loads -> one b128 LDS store
      #pragma unroll
      for (int it = 0; it < 8; ++it) {
        int n  = t & 63;
        int kb = (t >> 6) * 8 + it * 16;
        __bf16* dp = &Bs[n * LDA + kb];
        #pragma unroll
        for (int j = 0; j < 8; ++j)
          dp[j] = f2bf(W[(size_t)(k0 + kb + j) * 64 + n]);
      }
    } else {
      // ---- guarded tail chunk (zero pad K remainder) ----
      for (int it = 0; it < 16; ++it) {
        int r = it * 4 + (t >> 5);
        int c = (t & 31) * 4;
        int rg = row0 + r; if (rg > N_NODES - 1) rg = N_NODES - 1;
        __bf16* dp = &As[r * LDA + c];
        for (int j = 0; j < 4; ++j) {
          int kg = k0 + c + j;
          dp[j] = f2bf(kg < K ? A[(size_t)rg * K + kg] : 0.0f);
        }
      }
      for (int it = 0; it < 8; ++it) {
        int n  = t & 63;
        int kb = (t >> 6) * 8 + it * 16;
        __bf16* dp = &Bs[n * LDA + kb];
        for (int j = 0; j < 8; ++j) {
          int kg = k0 + kb + j;
          dp[j] = f2bf(kg < K ? W[(size_t)kg * 64 + n] : 0.0f);
        }
      }
    }
    __syncthreads();

    #pragma unroll
    for (int kk = 0; kk < KCHUNK; kk += 32) {
      // B 32x16 bf16, contiguous k order used here (matches A): lanes0-15 K=0..15; lanes16-31 K=16..31 (col = lane%16)
      v16bf b;
      const __bf16* bp = &Bs[ncol * LDA + kk + hi * 16];
      #pragma unroll
      for (int j = 0; j < 16; ++j) b[j] = bp[j];
      // A 16x32 bf16, contiguous k order used here (matches B): lanes0-15 K=0..15; lanes16-31 K=16..31 (row = lane%16)
      #pragma unroll
      for (int rt = 0; rt < 4; ++rt) {
        v16bf a;
        const __bf16* ap = &As[(rt * 16 + m) * LDA + kk + hi * 16];
        #pragma unroll
        for (int j = 0; j < 8; ++j) { a[j] = ap[j]; a[j + 8] = ap[8 + j]; }
        v8f& acc = (rt == 0) ? acc0 : (rt == 1) ? acc1 : (rt == 2) ? acc2 : acc3;
        acc = __builtin_amdgcn_wmma_f32_16x16x32_bf16(false, a, false, b,
                                                      (short)0, acc, false, false);
      }
    }
    __syncthreads();
  }

  // C/D layout: lanes0-15 VGPR v -> M=v; lanes16-31 -> M=8+v; N = lane%16
  const int n  = lane & 15;
  const int mb = (lane >> 4) * 8;
  #pragma unroll
  for (int rt = 0; rt < 4; ++rt) {
    const v8f& acc = (rt == 0) ? acc0 : (rt == 1) ? acc1 : (rt == 2) ? acc2 : acc3;
    int rowb = row0 + rt * 16 + mb;
    #pragma unroll
    for (int v = 0; v < 8; ++v) {
      int r = rowb + v;
      if (r < N_NODES)
        out[(size_t)r * 64 + (wave << 4) + n] = acc[v];
    }
  }
}

// ---------- per-node attention logits ----------
__global__ void att_kernel(const float* __restrict__ h, const float* __restrict__ att_src,
                           const float* __restrict__ att_dst,
                           float* __restrict__ a_src, float* __restrict__ a_dst, int H) {
  int n = blockIdx.x * blockDim.x + threadIdx.x;
  if (n >= N_NODES) return;
  const int C = 64 / H;
  float as[4] = {0,0,0,0}, ad[4] = {0,0,0,0};
  for (int c = 0; c < 64; ++c) {
    float v = h[(size_t)n * 64 + c];
    int hd = c / C;
    as[hd] += v * att_src[c];
    ad[hd] += v * att_dst[c];
  }
  for (int hd = 0; hd < H; ++hd) { a_src[n * H + hd] = as[hd]; a_dst[n * H + hd] = ad[hd]; }
}

// ---------- segment max over dst (uint-key atomicMax) ----------
__global__ void edge_max_kernel(const int* __restrict__ ei, const float* __restrict__ a_src,
                                const float* __restrict__ a_dst,
                                unsigned* __restrict__ segmax, int H) {
  long long gid = (long long)blockIdx.x * blockDim.x + threadIdx.x;
  int e = (int)(gid / H), hd = (int)(gid % H);
  if (e >= E_TOT) return;
  int s, d; edge_pair(ei, e, s, d);
  float l = a_src[s * H + hd] + a_dst[d * H + hd];
  l = l > 0.0f ? l : NEG_SLOPE * l;
  atomicMax(&segmax[d * H + hd], fkey(l));
}

// ---------- w = exp(lrelu - max); denom += w ----------
__global__ void edge_exp_kernel(const int* __restrict__ ei, const float* __restrict__ a_src,
                                const float* __restrict__ a_dst,
                                const unsigned* __restrict__ segmax,
                                float* __restrict__ wbuf, float* __restrict__ denom, int H) {
  long long gid = (long long)blockIdx.x * blockDim.x + threadIdx.x;
  int e = (int)(gid / H), hd = (int)(gid % H);
  if (e >= E_TOT) return;
  int s, d; edge_pair(ei, e, s, d);
  float l = a_src[s * H + hd] + a_dst[d * H + hd];
  l = l > 0.0f ? l : NEG_SLOPE * l;
  float w = __expf(l - funkey(segmax[d * H + hd]));
  wbuf[(size_t)e * H + hd] = w;
  atomicAdd(&denom[d * H + hd], w);
}

// ---------- scatter messages: accum[dst,c] += alpha * h[src,c] ----------
__global__ void edge_scatter_kernel(const int* __restrict__ ei, const float* __restrict__ h,
                                    const float* __restrict__ wbuf,
                                    const float* __restrict__ denom,
                                    float* __restrict__ accum, int H) {
  long long gid = (long long)blockIdx.x * blockDim.x + threadIdx.x;
  int e = (int)(gid >> 6), c = (int)(gid & 63);
  if (e >= E_TOT) return;
  int s, d; edge_pair(ei, e, s, d);
  const int C = 64 / H;
  int hd = c / C;
  float alpha = wbuf[(size_t)e * H + hd] / (denom[d * H + hd] + 1e-16f);
  atomicAdd(&accum[(size_t)d * 64 + c], alpha * h[(size_t)s * 64 + c]);
}

// ---------- BN column stats (sum, sumsq) ----------
__global__ void bnstats_kernel(const float* __restrict__ m, const float* __restrict__ bias,
                               float* __restrict__ stats) {
  int c  = threadIdx.x & 63;
  int rb = threadIdx.x >> 6;    // 0..3
  float s = 0.0f, s2 = 0.0f;
  for (int n = blockIdx.x * 4 + rb; n < N_NODES; n += gridDim.x * 4) {
    float v = m[(size_t)n * 64 + c] + bias[c];
    s += v; s2 += v * v;
  }
  atomicAdd(&stats[c], s);
  atomicAdd(&stats[64 + c], s2);
}

// ---------- BN apply ----------
__global__ void bnapply_kernel(const float* __restrict__ m, const float* __restrict__ bias,
                               const float* __restrict__ gamma, const float* __restrict__ beta,
                               const float* __restrict__ stats, float* __restrict__ out) {
  long long gid = (long long)blockIdx.x * blockDim.x + threadIdx.x;
  if (gid >= (long long)N_NODES * 64) return;
  int c = (int)(gid & 63);
  float mean = stats[c] * (1.0f / N_NODES);
  float var  = stats[64 + c] * (1.0f / N_NODES) - mean * mean;
  float v = m[gid] + bias[c];
  out[gid] = (v - mean) * rsqrtf(var + BN_EPS) * gamma[c] + beta[c];
}

// ---------- host ----------
static inline int blks(long long n, int b) { return (int)((n + b - 1) / b); }

extern "C" void kernel_launch(void* const* d_in, const int* in_sizes, int n_in,
                              void* d_out, int out_size, void* d_ws, size_t ws_size,
                              hipStream_t stream) {
  const float* x        = (const float*)d_in[0];
  const int*   ei       = (const int*)  d_in[1];   // edge_index [2, E] (int32: jax x64 off)
  const float* W1       = (const float*)d_in[2];
  const float* att_src1 = (const float*)d_in[3];
  const float* att_dst1 = (const float*)d_in[4];
  const float* b1       = (const float*)d_in[5];
  const float* gamma1   = (const float*)d_in[6];
  const float* beta1    = (const float*)d_in[7];
  const float* W2       = (const float*)d_in[8];
  const float* att_src2 = (const float*)d_in[9];
  const float* att_dst2 = (const float*)d_in[10];
  const float* b2       = (const float*)d_in[11];
  const float* gamma2   = (const float*)d_in[12];
  const float* beta2    = (const float*)d_in[13];
  float* out = (float*)d_out;

  // workspace carve (aliased across layers): ~70 MB
  float* h_lin  = (float*)d_ws;             // h1lin, later h2lin   [N*64]
  float* accum  = h_lin  + 3200000;         // accum1, later accum2 [N*64]
  float* h1bn   = accum  + 3200000;         //                      [N*64]
  float* wbuf   = h1bn   + 3200000;         // per-edge weights     [E_TOT*4]
  float* a_src1 = wbuf   + (size_t)E_TOT * 4;
  float* a_dst1 = a_src1 + N_NODES * 4;
  unsigned* segmax1 = (unsigned*)(a_dst1 + N_NODES * 4);
  float* denom1 = (float*)(segmax1 + N_NODES * 4);
  float* a_src2 = denom1 + N_NODES * 4;
  float* a_dst2 = a_src2 + N_NODES;
  unsigned* segmax2 = (unsigned*)(a_dst2 + N_NODES);
  float* denom2 = (float*)(segmax2 + N_NODES);
  float* stats  = denom2 + N_NODES;         // [128], reused per layer

  const long long NE = (long long)N_NODES * 64;
  const int gemm_blocks = (N_NODES + BM - 1) / BM;   // 782

  // ===== Layer 1 =====
  gemm_bf16_wmma<<<gemm_blocks, 128, 0, stream>>>(x, W1, h_lin, IN_F);
  att_kernel<<<blks(N_NODES, 256), 256, 0, stream>>>(h_lin, att_src1, att_dst1,
                                                     a_src1, a_dst1, 4);
  zero_kernel<<<blks(N_NODES * 4, 256), 256, 0, stream>>>((float*)segmax1, N_NODES * 4);
  zero_kernel<<<blks(N_NODES * 4, 256), 256, 0, stream>>>(denom1, N_NODES * 4);
  zero_kernel<<<blks(NE, 256), 256, 0, stream>>>(accum, NE);
  edge_max_kernel<<<blks((long long)E_TOT * 4, 256), 256, 0, stream>>>(ei, a_src1, a_dst1, segmax1, 4);
  edge_exp_kernel<<<blks((long long)E_TOT * 4, 256), 256, 0, stream>>>(ei, a_src1, a_dst1, segmax1,
                                                                       wbuf, denom1, 4);
  edge_scatter_kernel<<<blks((long long)E_TOT * 64, 256), 256, 0, stream>>>(ei, h_lin, wbuf,
                                                                            denom1, accum, 4);
  zero_kernel<<<1, 128, 0, stream>>>(stats, 128);
  bnstats_kernel<<<256, 256, 0, stream>>>(accum, b1, stats);
  bnapply_kernel<<<blks(NE, 256), 256, 0, stream>>>(accum, b1, gamma1, beta1, stats, h1bn);

  // ===== Layer 2 =====
  gemm_bf16_wmma<<<gemm_blocks, 128, 0, stream>>>(h1bn, W2, h_lin, OUTF);
  att_kernel<<<blks(N_NODES, 256), 256, 0, stream>>>(h_lin, att_src2, att_dst2,
                                                     a_src2, a_dst2, 1);
  zero_kernel<<<blks(N_NODES, 256), 256, 0, stream>>>((float*)segmax2, N_NODES);
  zero_kernel<<<blks(N_NODES, 256), 256, 0, stream>>>(denom2, N_NODES);
  zero_kernel<<<blks(NE, 256), 256, 0, stream>>>(accum, NE);
  edge_max_kernel<<<blks((long long)E_TOT, 256), 256, 0, stream>>>(ei, a_src2, a_dst2, segmax2, 1);
  edge_exp_kernel<<<blks((long long)E_TOT, 256), 256, 0, stream>>>(ei, a_src2, a_dst2, segmax2,
                                                                   wbuf, denom2, 1);
  edge_scatter_kernel<<<blks((long long)E_TOT * 64, 256), 256, 0, stream>>>(ei, h_lin, wbuf,
                                                                            denom2, accum, 1);
  zero_kernel<<<1, 128, 0, stream>>>(stats, 128);
  bnstats_kernel<<<256, 256, 0, stream>>>(accum, b2, stats);
  bnapply_kernel<<<blks(NE, 256), 256, 0, stream>>>(accum, b2, gamma2, beta2, stats, out);
}
